// EGNN_Spherical_51969104282141
// MI455X (gfx1250) — matrix operand order repaired and checked by reading, not compiled, not run on hardware
//
#include <hip/hip_runtime.h>

typedef __attribute__((ext_vector_type(16))) _Float16 v16h;
typedef __attribute__((ext_vector_type(8)))  float    v8f;

#define EDGES  160000
#define NNODES 10000
#define NRBF   50
#define EPSB   1e-5f

// A_P1 = 1/sqrt(2*32) = 0.125 ; A_P2 = A_P4 = 1/sqrt(2*16) ; A_P3 = 0.125
#define A_P1C 0.125f
#define A_P2C 0.17677669529663687f
#define A_P3C 0.125f
#define A_P4C 0.17677669529663687f
#define INV_SQRT3C 0.5773502691896258f

__device__ __forceinline__ float silu_f(float x) { return x / (1.0f + __expf(-x)); }

// Build 16x32 f16 A-fragment from an LDS tile (16 rows x 64 cols, row-major),
// following CDNA5 ISA 7.12.2: lane<16 -> K groups {kb+0..7, kb+16..23},
// lane>=16 -> K groups {kb+8..15, kb+24..31}; row M = lane&15.
__device__ __forceinline__ v16h build_a_frag(const _Float16* X, int lane, int kb) {
  const int m  = lane & 15;
  const int b0 = kb + ((lane < 16) ? 0 : 8);
  const int b1 = kb + ((lane < 16) ? 16 : 24);
  v16h a;
#pragma unroll
  for (int j = 0; j < 8; ++j) {
    a[j]     = X[m * 64 + b0 + j];
    a[8 + j] = X[m * 64 + b1 + j];
  }
  return a;
}

// Pre-swizzle an f32 weight matrix (Kin x Ncols) into f16 B-fragment layout:
// element idx = ((T*2+ks)*32 + lane)*16 + j  <->  k = ks*32 + (lane<16?0:8) + (j&7) + (j>=8?16:0),
// n = T*16 + (lane&15). Zero-pads K to 64. Each lane's fragment is then one
// contiguous 32-byte load (2x global_load_b128).
__global__ void swizzle_w_kernel(const float* __restrict__ W, _Float16* __restrict__ dst,
                                 int Kin, int Ncols, int total) {
  int idx = blockIdx.x * blockDim.x + threadIdx.x;
  if (idx >= total) return;
  const int j    = idx & 15;
  const int lane = (idx >> 4) & 31;
  const int ks   = (idx >> 9) & 1;
  const int T    = idx >> 10;
  const int k    = ks * 32 + ((lane < 16) ? 0 : 8) + (j & 7) + ((j >= 8) ? 16 : 0);
  const int nn   = T * 16 + (lane & 15);
  float v = (k < Kin && nn < Ncols) ? W[k * Ncols + nn] : 0.0f;
  dst[idx] = (_Float16)v;
}

__global__ void zero_kernel(float* p, int n) {
  int i = blockIdx.x * blockDim.x + threadIdx.x;
  if (i < n) p[i] = 0.0f;
}

// Edge MLP layers 1+2 (silu), f16 WMMA, K padded 50->64. One wave = 16 edges.
__global__ void __launch_bounds__(256)
edge_mlp12_kernel(const float* __restrict__ ef,
                  const _Float16* __restrict__ W1f,
                  const _Float16* __restrict__ W2f,
                  const float* __restrict__ b1,
                  const float* __restrict__ b2,
                  _Float16* __restrict__ Z2) {
  const int lane  = threadIdx.x & 31;
  const int wave  = threadIdx.x >> 5;
  const int ebase = (blockIdx.x * 8 + wave) * 16;
  __shared__ _Float16 sX[8][16 * 64];
  _Float16* X = sX[wave];

  for (int i = lane; i < 16 * 64; i += 32) {
    const int m = i >> 6, k = i & 63;
    const int e = ebase + m;
    float v = (k < NRBF && e < EDGES) ? ef[e * NRBF + k] : 0.0f;
    X[i] = (_Float16)v;
  }
  __syncthreads();

  v16h a0 = build_a_frag(X, lane, 0);
  v16h a1 = build_a_frag(X, lane, 32);
  const int n  = lane & 15;
  const int Mb = (lane < 16) ? 0 : 8;

  float z[4][8];
#pragma unroll
  for (int t = 0; t < 4; ++t) {
    v16h bA = *(const v16h*)(W1f + (t * 2 + 0) * 512 + lane * 16);
    v16h bB = *(const v16h*)(W1f + (t * 2 + 1) * 512 + lane * 16);
    v8f acc = {};
    acc = __builtin_amdgcn_wmma_f32_16x16x32_f16(false, a0, false, bA, (short)0, acc, false, false);
    acc = __builtin_amdgcn_wmma_f32_16x16x32_f16(false, a1, false, bB, (short)0, acc, false, false);
    const float bias = b1[t * 16 + n];
#pragma unroll
    for (int r = 0; r < 8; ++r) z[t][r] = silu_f(acc[r] + bias);
  }
  __syncthreads();
#pragma unroll
  for (int t = 0; t < 4; ++t)
#pragma unroll
    for (int r = 0; r < 8; ++r)
      X[(Mb + r) * 64 + t * 16 + n] = (_Float16)z[t][r];
  __syncthreads();

  a0 = build_a_frag(X, lane, 0);
  a1 = build_a_frag(X, lane, 32);
#pragma unroll
  for (int t = 0; t < 4; ++t) {
    v16h bA = *(const v16h*)(W2f + (t * 2 + 0) * 512 + lane * 16);
    v16h bB = *(const v16h*)(W2f + (t * 2 + 1) * 512 + lane * 16);
    v8f acc = {};
    acc = __builtin_amdgcn_wmma_f32_16x16x32_f16(false, a0, false, bA, (short)0, acc, false, false);
    acc = __builtin_amdgcn_wmma_f32_16x16x32_f16(false, a1, false, bB, (short)0, acc, false, false);
    const float bias = b2[t * 16 + n];
#pragma unroll
    for (int r = 0; r < 8; ++r) {
      const int e = ebase + Mb + r;
      if (e < EDGES) Z2[e * 64 + t * 16 + n] = (_Float16)silu_f(acc[r] + bias);
    }
  }
}

// Fused: w = silu2 @ W3 + b3 (16x2304 per wave, kept in WMMA accumulators only),
// immediately contracted with per-edge coefficients, scattered via atomics.
// Column segments: w1 [0,1024) u=T/2, w2 [1024,1280) u=T, w3 [1280,1792) u=T,
// w4 [1792,2304) u=T/2. One wave = 16 edges, 288 WMMA per wave.
// B fragments are software-pipelined: tile T+1's loads issue before tile T's
// WMMA + accumulation so global fetch overlaps matrix + VALU work.
__global__ void __launch_bounds__(128)
edge_contract_kernel(const _Float16* __restrict__ Z2,
                     const _Float16* __restrict__ W3f,
                     const float* __restrict__ b3,
                     const int* __restrict__ edge_index,
                     const float* __restrict__ edge_sh,
                     const float* __restrict__ h,
                     float* __restrict__ agg,
                     float* __restrict__ cnt) {
  const int lane  = threadIdx.x & 31;
  const int wave  = threadIdx.x >> 5;  // 4 waves
  const int ebase = (blockIdx.x * 4 + wave) * 16;

  __shared__ _Float16 sZ[4][1024];
  __shared__ float cw1[4][16][32];
  __shared__ float cw2[4][16][48];
  __shared__ float cw3[4][16][32];
  __shared__ float cw4[4][16][16];
  __shared__ float sy1[4][16][3];
  __shared__ int   sdst[4][16];

  _Float16* Z = sZ[wave];
  for (int i = lane; i < 1024; i += 32) {
    const int m = i >> 6, k = i & 63;
    const int e = ebase + m;
    Z[i] = (e < EDGES) ? Z2[e * 64 + k] : (_Float16)0.0f;
  }

  {
    const int m  = lane & 15;
    const int hf = lane >> 4;
    const int e  = ebase + m;
    const bool ok = e < EDGES;
    const int src = ok ? edge_index[e] : 0;
    const float y0 = ok ? edge_sh[e * 4 + 0] : 0.0f;
    const float yx = ok ? edge_sh[e * 4 + 1] : 0.0f;
    const float yy = ok ? edge_sh[e * 4 + 2] : 0.0f;
    const float yz = ok ? edge_sh[e * 4 + 3] : 0.0f;
    if (hf == 0) {
      sdst[wave][m] = ok ? edge_index[EDGES + e] : 0;
      sy1[wave][m][0] = yx; sy1[wave][m][1] = yy; sy1[wave][m][2] = yz;
    }
    const float* hs = h + (size_t)src * 80;
#pragma unroll
    for (int j = 0; j < 16; ++j) {
      const int u = hf * 16 + j;
      const float s = ok ? hs[u] : 0.0f;
      cw1[wave][m][u] = A_P1C * s * y0;   // coeff for w1 term -> m0
      cw3[wave][m][u] = A_P3C * s;        // coeff for w3 term -> m1 (x y1[i] at use)
    }
#pragma unroll
    for (int j = 0; j < 8; ++j) {
      const int u = hf * 8 + j;
      const float sx = ok ? hs[32 + u * 3 + 0] : 0.0f;
      const float sy = ok ? hs[32 + u * 3 + 1] : 0.0f;
      const float sz = ok ? hs[32 + u * 3 + 2] : 0.0f;
      cw2[wave][m][u * 3 + 0] = A_P2C * y0 * sx;
      cw2[wave][m][u * 3 + 1] = A_P2C * y0 * sy;
      cw2[wave][m][u * 3 + 2] = A_P2C * y0 * sz;
      cw4[wave][m][u] = A_P4C * INV_SQRT3C * (sx * yx + sy * yy + sz * yz);
    }
  }
  __syncthreads();

  const v16h a0 = build_a_frag(Z, lane, 0);
  const v16h a1 = build_a_frag(Z, lane, 32);
  const int n  = lane & 15;
  const int Mb = (lane < 16) ? 0 : 8;

  float m0a[2][8] = {};
  float m1a[3][8] = {};

  const _Float16* wp = W3f + lane * 16;  // per-lane fragment base
  v16h bA = *(const v16h*)(wp + 0);
  v16h bB = *(const v16h*)(wp + 512);

#pragma unroll 2
  for (int T = 0; T < 144; ++T) {
    // prefetch next tile's B fragments (dup last tile to stay in-bounds)
    const int Tn = (T + 1 < 144) ? T + 1 : T;
    const v16h nA = *(const v16h*)(wp + (Tn * 2 + 0) * 512);
    const v16h nB = *(const v16h*)(wp + (Tn * 2 + 1) * 512);

    v8f acc = {};
    acc = __builtin_amdgcn_wmma_f32_16x16x32_f16(false, a0, false, bA, (short)0, acc, false, false);
    acc = __builtin_amdgcn_wmma_f32_16x16x32_f16(false, a1, false, bB, (short)0, acc, false, false);
    const float bias = b3[T * 16 + n];

    if (T < 64) {                 // w1: col = u*32 + v
      const int u = T >> 1, p = T & 1;
#pragma unroll
      for (int r = 0; r < 8; ++r)
        m0a[p][r] += cw1[wave][Mb + r][u] * (acc[r] + bias);
    } else if (T < 80) {          // w2: col = 1024 + u*16 + w
      const int u = T - 64;
#pragma unroll
      for (int r = 0; r < 8; ++r) {
        const float wv = acc[r] + bias;
        const int M = Mb + r;
        m1a[0][r] += cw2[wave][M][u * 3 + 0] * wv;
        m1a[1][r] += cw2[wave][M][u * 3 + 1] * wv;
        m1a[2][r] += cw2[wave][M][u * 3 + 2] * wv;
      }
    } else if (T < 112) {         // w3: col = 1280 + u*16 + w
      const int u = T - 80;
#pragma unroll
      for (int r = 0; r < 8; ++r) {
        const int M = Mb + r;
        const float c = cw3[wave][M][u] * (acc[r] + bias);
        m1a[0][r] += c * sy1[wave][M][0];
        m1a[1][r] += c * sy1[wave][M][1];
        m1a[2][r] += c * sy1[wave][M][2];
      }
    } else {                      // w4: col = 1792 + u*32 + v
      const int tt = T - 112;
      const int u = tt >> 1, p = tt & 1;
#pragma unroll
      for (int r = 0; r < 8; ++r)
        m0a[p][r] += cw4[wave][Mb + r][u] * (acc[r] + bias);
    }
    bA = nA;
    bB = nB;
  }

#pragma unroll
  for (int r = 0; r < 8; ++r) {
    const int M = Mb + r;
    const int e = ebase + M;
    if (e < EDGES) {
      float* row = agg + (size_t)sdst[wave][M] * 80;
      atomicAdd(row + n,              m0a[0][r]);
      atomicAdd(row + 16 + n,         m0a[1][r]);
      atomicAdd(row + 32 + n * 3 + 0, m1a[0][r]);
      atomicAdd(row + 32 + n * 3 + 1, m1a[1][r]);
      atomicAdd(row + 32 + n * 3 + 2, m1a[2][r]);
    }
  }
  if (lane < 16) {
    const int e = ebase + lane;
    if (e < EDGES) atomicAdd(cnt + sdst[wave][lane], 1.0f);
  }
}

// Per-node: mean-aggregate + self interaction; accumulate BN statistics.
__global__ void __launch_bounds__(128)
node_kernel(const float* __restrict__ h,
            const float* __restrict__ A0,
            const float* __restrict__ A1p,
            const float* __restrict__ agg,
            const float* __restrict__ cnt,
            float* __restrict__ out,
            float* __restrict__ stats) {
  __shared__ float sA0[32 * 32];
  __shared__ float sA1[16 * 16];
  for (int i = threadIdx.x; i < 1024; i += blockDim.x) sA0[i] = A0[i];
  for (int i = threadIdx.x; i < 256;  i += blockDim.x) sA1[i] = A1p[i];
  __syncthreads();
  const int nid = blockIdx.x * blockDim.x + threadIdx.x;
  if (nid >= NNODES) return;
  const float invc = 1.0f / fmaxf(cnt[nid], 1.0f);
  const float* hn = h + (size_t)nid * 80;
  const float* ag = agg + (size_t)nid * 80;
  float* on = out + (size_t)nid * 80;
  const float inv_s0 = 0.17677669529663687f;  // 1/sqrt(32)
  const float inv_s1 = 0.25f;                 // 1/sqrt(16)
#pragma unroll 4
  for (int w = 0; w < 32; ++w) {
    float s = 0.0f;
    for (int u = 0; u < 32; ++u) s += hn[u] * sA0[u * 32 + w];
    const float x = ag[w] * invc + s * inv_s0;
    on[w] = x;
    atomicAdd(stats + w, x);
    atomicAdd(stats + 32 + w, x * x);
  }
#pragma unroll 2
  for (int w = 0; w < 16; ++w) {
    float nn = 0.0f;
    for (int i = 0; i < 3; ++i) {
      float s = 0.0f;
      for (int u = 0; u < 16; ++u) s += hn[32 + u * 3 + i] * sA1[u * 16 + w];
      const float x = ag[32 + w * 3 + i] * invc + s * inv_s1;
      on[32 + w * 3 + i] = x;
      nn += x * x;
    }
    atomicAdd(stats + 64 + w, nn);
  }
}

// BN normalization + residual, in-place on d_out.
__global__ void __launch_bounds__(256)
finalize_kernel(const float* __restrict__ h,
                const float* __restrict__ stats,
                const float* __restrict__ bn_w0,
                const float* __restrict__ bn_b0,
                const float* __restrict__ bn_w1,
                float* __restrict__ out) {
  const int nid = blockIdx.x * blockDim.x + threadIdx.x;
  if (nid >= NNODES) return;
  const float invN = 1.0f / (float)NNODES;
  const float* hn = h + (size_t)nid * 80;
  float* on = out + (size_t)nid * 80;
#pragma unroll 4
  for (int w = 0; w < 32; ++w) {
    const float mu  = stats[w] * invN;
    const float var = stats[32 + w] * invN - mu * mu;
    const float x = on[w];
    on[w] = (x - mu) * rsqrtf(var + EPSB) * bn_w0[w] + bn_b0[w] + hn[w];
  }
#pragma unroll 2
  for (int w = 0; w < 16; ++w) {
    const float nrm2 = stats[64 + w] * invN;
    const float sc = rsqrtf(nrm2 + EPSB) * bn_w1[w];
    for (int i = 0; i < 3; ++i) {
      const int c = 32 + w * 3 + i;
      on[c] = on[c] * sc + hn[c];
    }
  }
}

extern "C" void kernel_launch(void* const* d_in, const int* in_sizes, int n_in,
                              void* d_out, int out_size, void* d_ws, size_t ws_size,
                              hipStream_t stream) {
  (void)in_sizes; (void)n_in; (void)out_size; (void)ws_size;
  const float* h     = (const float*)d_in[0];
  const int*   eidx  = (const int*)d_in[1];
  const float* esh   = (const float*)d_in[2];
  const float* ef    = (const float*)d_in[3];
  const float* W1e   = (const float*)d_in[4];
  const float* b1e   = (const float*)d_in[5];
  const float* W2e   = (const float*)d_in[6];
  const float* b2e   = (const float*)d_in[7];
  const float* W3e   = (const float*)d_in[8];
  const float* b3e   = (const float*)d_in[9];
  const float* A0    = (const float*)d_in[10];
  const float* A1p   = (const float*)d_in[11];
  const float* bn_w0 = (const float*)d_in[12];
  const float* bn_b0 = (const float*)d_in[13];
  const float* bn_w1 = (const float*)d_in[14];
  float* out = (float*)d_out;

  char* ws = (char*)d_ws;
  _Float16* W1f = (_Float16*)(ws + 0);        // 4096 halves
  _Float16* W2f = (_Float16*)(ws + 8192);     // 4096 halves
  _Float16* W3f = (_Float16*)(ws + 16384);    // 147456 halves
  _Float16* Z2  = (_Float16*)(ws + 311296);   // E*64 halves = 20.48 MB
  float* agg    = (float*)(ws + 20791296);    // N*80 f32
  float* cnt    = agg + (size_t)NNODES * 80;  // N f32
  float* stats  = cnt + NNODES;               // 80 f32

  swizzle_w_kernel<<<(4096 + 255) / 256, 256, 0, stream>>>(W1e, W1f, 50, 64, 4096);
  swizzle_w_kernel<<<(4096 + 255) / 256, 256, 0, stream>>>(W2e, W2f, 64, 64, 4096);
  swizzle_w_kernel<<<(147456 + 255) / 256, 256, 0, stream>>>(W3e, W3f, 64, 2304, 147456);

  const int zn = NNODES * 81 + 80;  // agg + cnt + stats, contiguous
  zero_kernel<<<(zn + 255) / 256, 256, 0, stream>>>(agg, zn);

  edge_mlp12_kernel<<<(EDGES + 127) / 128, 256, 0, stream>>>(ef, W1f, W2f, b1e, b2e, Z2);
  edge_contract_kernel<<<(EDGES + 63) / 64, 128, 0, stream>>>(Z2, W3f, b3e, eidx, esh, h, agg, cnt);
  node_kernel<<<(NNODES + 127) / 128, 128, 0, stream>>>(h, A0, A1p, agg, cnt, out, stats);
  finalize_kernel<<<(NNODES + 255) / 256, 256, 0, stream>>>(h, stats, bn_w0, bn_b0, bn_w1, out);
}
